// Sequence_10462540333524
// MI455X (gfx1250) — matrix operand order repaired and checked by reading, not compiled, not run on hardware
//
#include <hip/hip_runtime.h>
#include <math.h>

// 2-layer LSTM (H=51) scan, B=512, T=512.
// Each workgroup owns 16 batch rows and scans all T steps with weights,
// state, and the input tile resident in LDS. Recurrent GEMMs run on
// v_wmma_f32_16x16x32_f16 (M=16 rows, N=204 gate cols in 13 tiles of 16
// statically distributed over 8 waves; K padded 51->64 / 102->128). Bias
// and the rank-1 x-term ride in the K padding rows of the WMMA, so the
// elementwise phases are pure activation math (v_exp/v_rcp, no divides).

typedef __attribute__((ext_vector_type(16))) _Float16 v16h;
typedef __attribute__((ext_vector_type(8)))  _Float16 v8h;
typedef __attribute__((ext_vector_type(8)))  float    v8f;

namespace {

constexpr int H  = 51;
constexpr int G  = 4 * H;      // 204 gate columns
constexpr int B  = 512;
constexpr int T  = 512;
constexpr int ZS = 208;        // sZ column stride (floats)
constexpr int W1S = 72;        // sW1 K-stride in halves (conflict-free: 36 dwords)
constexpr int W2S = 136;       // sW2 K-stride in halves (68 dwords)
constexpr int AS  = 72;        // h-state (A matrix) K-stride in halves
constexpr int CS  = 52;        // f32 state row stride

// byte offsets into dynamic LDS (all 16B aligned)
constexpr int OFF_W1 = 0;                        // 208*72*2  = 29952
constexpr int OFF_W2 = OFF_W1 + 208 * W1S * 2;   // 208*136*2 = 56576
constexpr int OFF_Z  = OFF_W2 + 208 * W2S * 2;   // 16*208*4  = 13312
constexpr int OFF_A1 = OFF_Z  + 16 * ZS * 4;     // 16*72*2   = 2304
constexpr int OFF_A2 = OFF_A1 + 16 * AS * 2;     // 2304
constexpr int OFF_C1 = OFF_A2 + 16 * AS * 2;     // 16*52*4   = 3328
constexpr int OFF_C2 = OFF_C1 + 16 * CS * 4;     // 3328
constexpr int OFF_H2 = OFF_C2 + 16 * CS * 4;     // 3328
constexpr int OFF_WL = OFF_H2 + 16 * CS * 4;     // 64*4      = 256
constexpr int OFF_X  = OFF_WL + 256;             // 16*512*4  = 32768
constexpr int SMEM_BYTES = OFF_X + 16 * T * 4;   // 147456 B  (< 320KB WGP LDS)

__device__ __forceinline__ float sigmoid_fast(float x) {
  // v_exp_f32 + v_rcp_f32; saturates correctly at both ends.
  return __builtin_amdgcn_rcpf(1.0f + __expf(-x));
}

__device__ __forceinline__ float tanh_fast(float x) {
  // tanh(|x|) = (1-e)/(1+e), e = exp(-2|x|) in (0,1] -> no overflow; restore sign.
  float ax = __builtin_fabsf(x);
  float e  = __expf(-2.0f * ax);
  float r  = (1.0f - e) * __builtin_amdgcn_rcpf(1.0f + e);
  return __builtin_copysignf(r, x);
}

__device__ __forceinline__ v8f wmma_f16(v16h a, v16h b, v8f c) {
  return __builtin_amdgcn_wmma_f32_16x16x32_f16(
      /*neg_a=*/false, a, /*neg_b=*/false, b,
      /*c_mod=*/(short)0, c, /*reuse_a=*/false, /*reuse_b=*/false);
}

// A fragment (16x32 f16) from row-major [m][k] LDS, K-stride AS halves.
// lanes 0-15: m=lane, V0-3 = K(kc+0..7),  V4-7 = K(kc+16..23)
// lanes 16-31: m=lane-16, V0-3 = K(kc+8..15), V4-7 = K(kc+24..31)
__device__ __forceinline__ v16h load_A(const _Float16* base, int lane, int kc) {
  const int m  = lane & 15;
  const int h8 = (lane >> 4) * 8;
  const _Float16* p = base + m * AS + kc + h8;   // 16B aligned (AS*2 = 144 = 9*16)
  v8h lo = *(const v8h*)(p);
  v8h hi = *(const v8h*)(p + 16);
  return __builtin_shufflevector(lo, hi, 0, 1, 2, 3, 4, 5, 6, 7,
                                 8, 9, 10, 11, 12, 13, 14, 15);
}

// B fragment (32x16 f16) from [n][k] LDS weight array (W[n][k] = B^T).
// lanes 0-15: n=lane, K=kc+0..7,kc+16..23; lanes 16-31: n=lane-16, K=kc+8..15,kc+24..31.
__device__ __forceinline__ v16h load_B(const _Float16* base, int stride,
                                       int lane, int tile, int kc) {
  const int n  = lane & 15;
  const int kb = (lane >> 4) * 8;
  const _Float16* p = base + (tile * 16 + n) * stride + kc + kb;
  v8h lo = *(const v8h*)(p);
  v8h hi = *(const v8h*)(p + 16);
  return __builtin_shufflevector(lo, hi, 0, 1, 2, 3, 4, 5, 6, 7,
                                 8, 9, 10, 11, 12, 13, 14, 15);
}

// C layout: VGPR r -> lanes 0-15: M=r, N=lane; lanes 16-31: M=r+8, N=lane-16.
__device__ __forceinline__ void store_C(float* z, int lane, int tile, v8f c) {
  const int n  = tile * 16 + (lane & 15);
  const int mo = (lane >> 4) * 8;
#pragma unroll
  for (int r = 0; r < 8; ++r) z[(mo + r) * ZS + n] = c[r];
}

} // namespace

__global__ __launch_bounds__(256, 1) void lstm_scan_kernel(
    const float* __restrict__ input,   // [B, T]
    const float* __restrict__ W_ih1,   // [4H, 1]
    const float* __restrict__ W_hh1,   // [4H, H]
    const float* __restrict__ b_ih1,   // [4H]
    const float* __restrict__ b_hh1,   // [4H]
    const float* __restrict__ W_ih2,   // [4H, H]
    const float* __restrict__ W_hh2,   // [4H, H]
    const float* __restrict__ b_ih2,   // [4H]
    const float* __restrict__ b_hh2,   // [4H]
    const float* __restrict__ W_lin,   // [1, H]
    const float* __restrict__ b_lin,   // [1]
    float* __restrict__ out)           // [B, T]
{
  extern __shared__ unsigned char smem[];
  _Float16* sW1 = (_Float16*)(smem + OFF_W1);   // [208][72]  Whh1 + bias row(51) + Wih1 row(52)
  _Float16* sW2 = (_Float16*)(smem + OFF_W2);   // [208][136] Wih2(K0-50) + bias2 row(51) + Whh2(K64-114)
  float*    sZ  = (float*)(smem + OFF_Z);       // [16][208]  gate pre-activations
  _Float16* sA1 = (_Float16*)(smem + OFF_A1);   // [16][72]   h1 f16; row51=1.0, row52=x_t
  _Float16* sA2 = (_Float16*)(smem + OFF_A2);   // [16][72]   h2 f16
  float*    sC1 = (float*)(smem + OFF_C1);      // [16][52]   c1
  float*    sC2 = (float*)(smem + OFF_C2);      // [16][52]   c2
  float*    sH2 = (float*)(smem + OFF_H2);      // [16][52]   h2 f32 (projection; [.][51] = 0 pad)
  float*    sWl = (float*)(smem + OFF_WL);      // [52] (element 51 = 0 pad)
  float*    sX  = (float*)(smem + OFF_X);       // [16][512]  whole input tile

  const int tid  = threadIdx.x;
  const int lane = tid & 31;
  // wave index as an SGPR -> uniform (scalar) control flow for tile loops
  const int wv   = __builtin_amdgcn_readfirstlane(tid >> 5);
  const int row0 = blockIdx.x * 16;

  // Static tile assignment over the 13 N-tiles. Wave 0 also owns the
  // phase-E projection (which overlaps the next phase A), so it gets 1 tile:
  //   wave:  0  1  2  3  4  5  6  7
  //   start: 0  1  3  5  7  9 11 12
  //   count: 1  2  2  2  2  2  1  1
  const int tcnt = (wv >= 1 && wv <= 5) ? 2 : 1;
  const int tst  = (wv == 0) ? 0 : ((wv <= 5) ? 2 * wv - 1 : wv + 5);

  // ---- one-time staging -------------------------------------------------
  for (int idx = tid; idx < 208 * W1S; idx += 256) {
    const int n = idx / W1S, k = idx % W1S;
    float v = 0.0f;
    if (n < G) {
      if (k < H)        v = W_hh1[n * H + k];
      else if (k == 51) v = b_ih1[n] + b_hh1[n];  // pairs with A row51 == 1.0
      else if (k == 52) v = W_ih1[n];             // pairs with A row52 == x_t
    }
    sW1[idx] = (_Float16)v;
  }
  for (int idx = tid; idx < 208 * W2S; idx += 256) {
    const int n = idx / W2S, k = idx % W2S;
    float v = 0.0f;
    if (n < G) {
      if (k < H)                      v = W_ih2[n * H + k];        // K 0..50  <- h1
      else if (k == 51)               v = b_ih2[n] + b_hh2[n];     // <- A1 row51 == 1.0
      else if (k >= 64 && k < 64 + H) v = W_hh2[n * H + (k - 64)]; // K 64..114 <- h2
    }
    sW2[idx] = (_Float16)v;
  }
  for (int idx = tid; idx < 16 * AS; idx += 256) { sA1[idx] = (_Float16)0.0f; sA2[idx] = (_Float16)0.0f; }
  for (int idx = tid; idx < 16 * CS; idx += 256) { sC1[idx] = 0.0f; sC2[idx] = 0.0f; sH2[idx] = 0.0f; }
  for (int idx = tid; idx < 52; idx += 256) sWl[idx] = (idx < H) ? W_lin[idx] : 0.0f;
  for (int idx = tid; idx < 16 * T; idx += 256) {   // coalesced along T
    const int m = idx >> 9, t = idx & (T - 1);
    sX[idx] = input[(size_t)(row0 + m) * T + t];
  }
  const float blin = b_lin[0];
  __syncthreads();
  if (tid < 16) {
    sA1[tid * AS + 51] = (_Float16)1.0f;            // persistent bias row
    sA1[tid * AS + 52] = (_Float16)sX[tid * T + 0]; // x_0
  }
  __syncthreads();

  for (int t = 0; t < T; ++t) {
    // ---- phase A: z1 = [h1|1|x] @ [Whh1|b1|Wih1]^T  (16 x 204, K=64)
    {
      v16h a0 = load_A(sA1, lane, 0);
      v16h a1 = load_A(sA1, lane, 32);
      for (int i = 0; i < tcnt; ++i) {        // scalar trip count -> uniform CF
        const int tile = tst + i;
        v8f z = {};
        v8f acc0 = wmma_f16(a0, load_B(sW1, W1S, lane, tile, 0), z);
        v8f acc1 = wmma_f16(a1, load_B(sW1, W1S, lane, tile, 32), z);
        store_C(sZ, lane, tile, acc0 + acc1);
      }
    }
    __syncthreads();

    // ---- phase B: layer-1 activations (bias/x already in z)
    for (int idx = tid; idx < 16 * H; idx += 256) {
      const int m = (unsigned)idx / 51u;
      const int j = idx - m * 51;
      const float* zr = sZ + m * ZS;
      float ig = sigmoid_fast(zr[j]);
      float fg = sigmoid_fast(zr[H + j]);
      float gg = tanh_fast   (zr[2 * H + j]);
      float og = sigmoid_fast(zr[3 * H + j]);
      float c  = __builtin_fmaf(fg, sC1[m * CS + j], ig * gg);
      sC1[m * CS + j] = c;
      sA1[m * AS + j] = (_Float16)(og * tanh_fast(c));
    }
    if (tid < 16) {                 // stage x_{t+1} into the A row for next step
      const int tn = (t + 1 < T) ? t + 1 : 0;
      sA1[tid * AS + 52] = (_Float16)sX[tid * T + tn];
    }
    __syncthreads();

    // ---- phase C: z2 = [h1|1|..|h2] @ [Wih2|b2|..|Whh2]^T  (16 x 204, K=128)
    {
      v16h a0 = load_A(sA1, lane, 0);
      v16h a1 = load_A(sA1, lane, 32);
      v16h a2 = load_A(sA2, lane, 0);
      v16h a3 = load_A(sA2, lane, 32);
      for (int i = 0; i < tcnt; ++i) {
        const int tile = tst + i;
        v8f z = {};
        // two independent accumulator chains -> adjacent WMMAs independent
        v8f acc0 = wmma_f16(a0, load_B(sW2, W2S, lane, tile, 0), z);
        v8f acc1 = wmma_f16(a1, load_B(sW2, W2S, lane, tile, 32), z);
        acc0 = wmma_f16(a2, load_B(sW2, W2S, lane, tile, 64), acc0);
        acc1 = wmma_f16(a3, load_B(sW2, W2S, lane, tile, 96), acc1);
        store_C(sZ, lane, tile, acc0 + acc1);
      }
    }
    __syncthreads();

    // ---- phase D: layer-2 activations
    for (int idx = tid; idx < 16 * H; idx += 256) {
      const int m = (unsigned)idx / 51u;
      const int j = idx - m * 51;
      const float* zr = sZ + m * ZS;
      float ig = sigmoid_fast(zr[j]);
      float fg = sigmoid_fast(zr[H + j]);
      float gg = tanh_fast   (zr[2 * H + j]);
      float og = sigmoid_fast(zr[3 * H + j]);
      float c  = __builtin_fmaf(fg, sC2[m * CS + j], ig * gg);
      sC2[m * CS + j] = c;
      float h  = og * tanh_fast(c);
      sA2[m * AS + j] = (_Float16)h;
      sH2[m * CS + j] = h;
    }
    __syncthreads();

    // ---- phase E: out[:, t] = h2 @ W_lin^T + b   (wave 0; overlaps next A)
    if (wv == 0) {
      const int m  = lane & 15;
      const int hh = lane >> 4;
      const int j0 = hh * 26;       // fixed 26-wide halves (padded with zeros)
      float acc = 0.0f;
#pragma unroll
      for (int j = 0; j < 26; ++j)
        acc = __builtin_fmaf(sH2[m * CS + j0 + j], sWl[j0 + j], acc);
      acc += __shfl_down(acc, 16);
      if (hh == 0) out[(size_t)(row0 + m) * T + t] = acc + blin;
    }
    // no barrier needed: next phase A touches only sZ/sA1, both already
    // barrier-separated from their last readers/writers.
  }
}

extern "C" void kernel_launch(void* const* d_in, const int* in_sizes, int n_in,
                              void* d_out, int out_size, void* d_ws, size_t ws_size,
                              hipStream_t stream) {
  (void)in_sizes; (void)n_in; (void)out_size; (void)d_ws; (void)ws_size;
  const float* input = (const float*)d_in[0];
  const float* W_ih1 = (const float*)d_in[1];
  const float* W_hh1 = (const float*)d_in[2];
  const float* b_ih1 = (const float*)d_in[3];
  const float* b_hh1 = (const float*)d_in[4];
  const float* W_ih2 = (const float*)d_in[5];
  const float* W_hh2 = (const float*)d_in[6];
  const float* b_ih2 = (const float*)d_in[7];
  const float* b_hh2 = (const float*)d_in[8];
  const float* W_lin = (const float*)d_in[9];
  const float* b_lin = (const float*)d_in[10];
  // d_in[11] = future (int scalar, reference default 0) -- unused

  dim3 grid(B / 16);   // 32 independent batch tiles
  dim3 block(256);     // 8 wave32 waves
  lstm_scan_kernel<<<grid, block, SMEM_BYTES, stream>>>(
      input, W_ih1, W_hh1, b_ih1, b_hh1,
      W_ih2, W_hh2, b_ih2, b_hh2, W_lin, b_lin, (float*)d_out);
}
